// VulnSage_50577534877709
// MI455X (gfx1250) — matrix operand order repaired and checked by reading, not compiled, not run on hardware
//
#include <hip/hip_runtime.h>
#include <hip/hip_bf16.h>
#include <math.h>

typedef _Float16 half_t;
typedef __attribute__((ext_vector_type(16))) _Float16 v16h;
typedef __attribute__((ext_vector_type(8)))  _Float16 v8h;
typedef __attribute__((ext_vector_type(8)))  float    v8f;
typedef __attribute__((ext_vector_type(4)))  int      v4i;

#define AS1 __attribute__((address_space(1)))
#define AS3 __attribute__((address_space(3)))

#define D_DIM 2048
#define B_DIM 4096
#define NT 4
#define NH 16
#define DK 128
#define M_DIM (B_DIM * NT)   // 16384

#define BM 128
#define BN 128
#define BK 64
#define NSTAGES (D_DIM / BK)         // 32
#define CHUNKS_PER_THREAD 8          // 4 (A) + 4 (B) async b128 issues / thread / stage

#if __has_builtin(__builtin_amdgcn_global_load_async_to_lds_b128)
#define HAVE_ASYNC 1
#else
#define HAVE_ASYNC 0
#endif

template <int N>
__device__ __forceinline__ void wait_async() {
#if __has_builtin(__builtin_amdgcn_s_wait_asynccnt)
    __builtin_amdgcn_s_wait_asynccnt(N);
#else
    asm volatile("s_wait_asynccnt %0" :: "n"(N) : "memory");
#endif
}

__device__ __forceinline__ void copy16_to_lds(const half_t* gsrc, half_t* ldst) {
#if HAVE_ASYNC
    __builtin_amdgcn_global_load_async_to_lds_b128((AS1 v4i*)gsrc, (AS3 v4i*)ldst, 0, 0);
#else
    *(v8h*)ldst = *(const v8h*)gsrc;
#endif
}

// ---------------------------------------------------------------------------
// f32 -> f16 elementwise conversion (weights)
// ---------------------------------------------------------------------------
__global__ __launch_bounds__(256) void cvt_f32_f16(const float* __restrict__ src,
                                                   half_t* __restrict__ dst, int n) {
    int i = blockIdx.x * 256 + threadIdx.x;
    if (i < n) dst[i] = (half_t)src[i];
}

// ---------------------------------------------------------------------------
// Build H (M_DIM x D) f16: row b*4+t = e_t[b]
// ---------------------------------------------------------------------------
__global__ __launch_bounds__(256) void build_h(const float* __restrict__ e0,
                                               const float* __restrict__ e1,
                                               const float* __restrict__ e2,
                                               const float* __restrict__ e3,
                                               half_t* __restrict__ H) {
    size_t i = (size_t)blockIdx.x * 256 + threadIdx.x;   // over B*D
    size_t b = i / D_DIM;
    size_t d = i % D_DIM;
    size_t base = (b * NT) * (size_t)D_DIM + d;
    H[base + 0 * D_DIM] = (half_t)e0[i];
    H[base + 1 * D_DIM] = (half_t)e1[i];
    H[base + 2 * D_DIM] = (half_t)e2[i];
    H[base + 3 * D_DIM] = (half_t)e3[i];
}

// ---------------------------------------------------------------------------
// WMMA GEMM: C[m,n] = sum_k A[m,k] * W[n,k] + bias[n]
//   A : (M_DIM x D) f16 row-major
//   W : (D x D)     f16 row-major (rows are the N/output dim -> B-tile is [n][k])
//   C : (M_DIM x D) OutT
// Block tile 128x128, 8 waves in 4(M) x 2(N) grid, 32x64 per wave = 2x4 WMMA
// accumulators (64 VGPRs -> clean allocation, no VGPR-MSB shuffling).
// BK=64: two WMMA K-halves per stage -> 16 WMMAs between barriers per wave.
// A/B staged in double-buffered LDS via async-to-LDS (ASYNCcnt).
// ---------------------------------------------------------------------------
template <typename OutT>
__global__ __launch_bounds__(256) void gemm_wmma(const half_t* __restrict__ A,
                                                 const half_t* __restrict__ W,
                                                 const float* __restrict__ bias,
                                                 OutT* __restrict__ C) {
    __shared__ __align__(16) half_t As[2][BM][BK];  // 2 x 16 KB
    __shared__ __align__(16) half_t Bs[2][BN][BK];  // 2 x 16 KB

    const int tid  = threadIdx.x;
    const int lane = tid & 31;
    const int wave = tid >> 5;      // 0..7
    const int wm   = wave >> 1;     // 0..3 : 32-row band
    const int wn   = wave & 1;      // 0..1 : 64-col band
    const int m0   = blockIdx.x * BM;
    const int n0   = blockIdx.y * BN;
    const int l16  = lane & 15;
    const int lhi  = lane >> 4;     // 0 or 1

    v8f acc[2][4] = {};

    union AV { v16h v; struct { v8h lo; v8h hi; } p; };

    // Stage one BK-slice of A and B into LDS buffer `buf`.
    auto stage = [&](int buf, int k0) {
#pragma unroll
        for (int i = 0; i < 4; ++i) {
            int chunk = tid + i * 256;                  // 0..1023
            int r  = chunk >> 3;                        // 0..127
            int kc = (chunk & 7) * 8;                   // 0,8,...,56
            copy16_to_lds(&A[(size_t)(m0 + r) * D_DIM + k0 + kc], &As[buf][r][kc]);
            copy16_to_lds(&W[(size_t)(n0 + r) * D_DIM + k0 + kc], &Bs[buf][r][kc]);
        }
    };

    stage(0, 0);

    for (int s = 0; s < NSTAGES; ++s) {
        const int buf = s & 1;
        if (s + 1 < NSTAGES) {
            stage(buf ^ 1, (s + 1) * BK);
#if HAVE_ASYNC
            wait_async<CHUNKS_PER_THREAD>();   // current stage landed; next in flight
#endif
        } else {
#if HAVE_ASYNC
            wait_async<0>();
#endif
        }
        __syncthreads();

#pragma unroll
        for (int kh = 0; kh < 2; ++kh) {          // two K=32 halves of the slab
            // A fragments (16x32): lanes 0-15 K{0..7,16..23}, lanes 16-31 K{8..15,24..31}
            AV a[2];
#pragma unroll
            for (int mi = 0; mi < 2; ++mi) {
                int r  = wm * 32 + mi * 16 + l16;
                int kb = kh * 32 + lhi * 8;
                a[mi].p.lo = *(const v8h*)&As[buf][r][kb];
                a[mi].p.hi = *(const v8h*)&As[buf][r][kb + 16];
            }
            // B fragments (32x16): lane<16 -> col N=lane, K{0..7,16..23}; lane>=16 -> K{8..15,24..31}
#pragma unroll
            for (int ni = 0; ni < 4; ++ni) {
                int n   = wn * 64 + ni * 16 + l16;
                int kb2 = kh * 32 + lhi * 8;
                AV b;
                b.p.lo = *(const v8h*)&Bs[buf][n][kb2];
                b.p.hi = *(const v8h*)&Bs[buf][n][kb2 + 16];
#pragma unroll
                for (int mi = 0; mi < 2; ++mi) {
                    acc[mi][ni] = __builtin_amdgcn_wmma_f32_16x16x32_f16(
                        false, a[mi].v, false, b.v, (short)0, acc[mi][ni], false, false);
                }
            }
        }
        __syncthreads();
    }

    // Epilogue: VGPR j -> (M=j, N=lane) for lanes 0-15, (M=j+8, N=lane-16) for 16-31
#pragma unroll
    for (int mi = 0; mi < 2; ++mi) {
#pragma unroll
        for (int ni = 0; ni < 4; ++ni) {
            int col = n0 + wn * 64 + ni * 16 + l16;
            float bv = bias[col];
#pragma unroll
            for (int j = 0; j < 8; ++j) {
                int row = m0 + wm * 32 + mi * 16 + lhi * 8 + j;
                C[(size_t)row * D_DIM + col] = (OutT)(acc[mi][ni][j] + bv);
            }
        }
    }
}

// ---------------------------------------------------------------------------
// Attention: one wave per (b,h). 4x4 scores, quality-gated softmax, ctx.
// Q/K/V layout: (M_DIM x D) f16, head h in columns [h*128, h*128+128).
// ---------------------------------------------------------------------------
__global__ __launch_bounds__(256) void attn_kernel(const half_t* __restrict__ Q,
                                                   const half_t* __restrict__ K,
                                                   const half_t* __restrict__ V,
                                                   const float* __restrict__ quality,
                                                   half_t* __restrict__ Ctx) {
    int idx  = blockIdx.x * 8 + (threadIdx.x >> 5);  // (b,h) pair, 0..65535
    int lane = threadIdx.x & 31;
    int b = idx >> 4;
    int h = idx & 15;
    const size_t base = (size_t)b * NT * D_DIM + (size_t)h * DK;

    float q[NT][4], k[NT][4], v[NT][4];
#pragma unroll
    for (int t = 0; t < NT; ++t) {
        size_t r = base + (size_t)t * D_DIM + lane;
#pragma unroll
        for (int c = 0; c < 4; ++c) {
            q[t][c] = (float)Q[r + 32 * c];
            k[t][c] = (float)K[r + 32 * c];
            v[t][c] = (float)V[r + 32 * c];
        }
    }

    float s[NT][NT];
#pragma unroll
    for (int i = 0; i < NT; ++i)
#pragma unroll
        for (int j = 0; j < NT; ++j) {
            float acc = 0.f;
#pragma unroll
            for (int c = 0; c < 4; ++c) acc += q[i][c] * k[j][c];
            for (int off = 16; off > 0; off >>= 1)
                acc += __shfl_xor(acc, off, 32);
            s[i][j] = acc;
        }

    const float scale = 0.08838834764831845f;  // 1/sqrt(128)
    float qual[NT];
#pragma unroll
    for (int j = 0; j < NT; ++j) qual[j] = quality[(size_t)b * NT + j];

    float p[NT][NT];
#pragma unroll
    for (int i = 0; i < NT; ++i) {
        float mx = -1e30f;
#pragma unroll
        for (int j = 0; j < NT; ++j) {
            s[i][j] = s[i][j] * scale * qual[j];
            mx = fmaxf(mx, s[i][j]);
        }
        float sum = 0.f;
#pragma unroll
        for (int j = 0; j < NT; ++j) {
            p[i][j] = __expf(s[i][j] - mx);
            sum += p[i][j];
        }
        float inv = 1.f / sum;
#pragma unroll
        for (int j = 0; j < NT; ++j) p[i][j] *= inv;
    }

#pragma unroll
    for (int i = 0; i < NT; ++i) {
        size_t r = base + (size_t)i * D_DIM + lane;
#pragma unroll
        for (int c = 0; c < 4; ++c) {
            float acc = 0.f;
#pragma unroll
            for (int j = 0; j < NT; ++j) acc += p[i][j] * v[j][c];
            Ctx[r + 32 * c] = (half_t)acc;
        }
    }
}

// ---------------------------------------------------------------------------
// Finalize: fused = mean_t ctxP, residual = mean_t H, LayerNorm -> out
// One block (256 threads) per batch row b; each thread owns 8 columns.
// ---------------------------------------------------------------------------
__global__ __launch_bounds__(256) void finalize_kernel(const float* __restrict__ ctxP,
                                                       const float* __restrict__ e0,
                                                       const float* __restrict__ e1,
                                                       const float* __restrict__ e2,
                                                       const float* __restrict__ e3,
                                                       const float* __restrict__ gamma,
                                                       const float* __restrict__ beta,
                                                       float* __restrict__ out) {
    int b   = blockIdx.x;
    int tid = threadIdx.x;
    __shared__ float red[256];

    float y[8];
    float sum = 0.f;
#pragma unroll
    for (int i = 0; i < 8; ++i) {
        int d = tid + i * 256;
        size_t r = (size_t)b * NT * D_DIM + d;
        float fused = 0.25f * (ctxP[r] + ctxP[r + D_DIM] + ctxP[r + 2 * D_DIM] + ctxP[r + 3 * D_DIM]);
        size_t g = (size_t)b * D_DIM + d;
        float resid = 0.25f * (e0[g] + e1[g] + e2[g] + e3[g]);
        y[i] = fused + resid;
        sum += y[i];
    }

    red[tid] = sum;
    __syncthreads();
    for (int s = 128; s > 0; s >>= 1) {
        if (tid < s) red[tid] += red[tid + s];
        __syncthreads();
    }
    float mu = red[0] * (1.f / D_DIM);
    __syncthreads();

    float sq = 0.f;
#pragma unroll
    for (int i = 0; i < 8; ++i) {
        float dl = y[i] - mu;
        sq += dl * dl;
    }
    red[tid] = sq;
    __syncthreads();
    for (int s = 128; s > 0; s >>= 1) {
        if (tid < s) red[tid] += red[tid + s];
        __syncthreads();
    }
    float rstd = rsqrtf(red[0] * (1.f / D_DIM) + 1e-5f);

#pragma unroll
    for (int i = 0; i < 8; ++i) {
        int d = tid + i * 256;
        out[(size_t)b * D_DIM + d] = (y[i] - mu) * rstd * gamma[d] + beta[d];
    }
}

// ---------------------------------------------------------------------------
extern "C" void kernel_launch(void* const* d_in, const int* in_sizes, int n_in,
                              void* d_out, int out_size, void* d_ws, size_t ws_size,
                              hipStream_t stream) {
    const float* e0      = (const float*)d_in[0];
    const float* e1      = (const float*)d_in[1];
    const float* e2      = (const float*)d_in[2];
    const float* e3      = (const float*)d_in[3];
    const float* quality = (const float*)d_in[4];
    const float* Wq      = (const float*)d_in[5];
    const float* bq      = (const float*)d_in[6];
    const float* Wk      = (const float*)d_in[7];
    const float* bk      = (const float*)d_in[8];
    const float* Wv      = (const float*)d_in[9];
    const float* bv      = (const float*)d_in[10];
    const float* Wo      = (const float*)d_in[11];
    const float* bo      = (const float*)d_in[12];
    const float* gamma   = (const float*)d_in[13];
    const float* beta    = (const float*)d_in[14];
    float* out = (float*)d_out;

    // Workspace layout
    char* ws = (char*)d_ws;
    size_t off = 0;
    auto alloc = [&](size_t bytes) { void* p = ws + off; off += (bytes + 255) & ~(size_t)255; return p; };

    const size_t WELEM = (size_t)D_DIM * D_DIM;          // 4,194,304
    const size_t MELEM = (size_t)M_DIM * D_DIM;          // 33,554,432

    half_t* Wqh  = (half_t*)alloc(WELEM * sizeof(half_t));
    half_t* Wkh  = (half_t*)alloc(WELEM * sizeof(half_t));
    half_t* Wvh  = (half_t*)alloc(WELEM * sizeof(half_t));
    half_t* Woh  = (half_t*)alloc(WELEM * sizeof(half_t));
    half_t* H16  = (half_t*)alloc(MELEM * sizeof(half_t));
    half_t* Qh   = (half_t*)alloc(MELEM * sizeof(half_t));
    half_t* Kh   = (half_t*)alloc(MELEM * sizeof(half_t));
    half_t* Vh   = (half_t*)alloc(MELEM * sizeof(half_t));
    half_t* Ctxh = (half_t*)alloc(MELEM * sizeof(half_t));
    float*  CtxP = (float*)alloc(MELEM * sizeof(float));

    // 1. Weight conversion
    {
        int n = (int)WELEM;
        int blocks = (n + 255) / 256;
        cvt_f32_f16<<<blocks, 256, 0, stream>>>(Wq, Wqh, n);
        cvt_f32_f16<<<blocks, 256, 0, stream>>>(Wk, Wkh, n);
        cvt_f32_f16<<<blocks, 256, 0, stream>>>(Wv, Wvh, n);
        cvt_f32_f16<<<blocks, 256, 0, stream>>>(Wo, Woh, n);
    }

    // 2. Build H
    {
        int blocks = (int)(((size_t)B_DIM * D_DIM) / 256);
        build_h<<<blocks, 256, 0, stream>>>(e0, e1, e2, e3, H16);
    }

    // 3. QKV GEMMs
    dim3 ggrid(M_DIM / BM, D_DIM / BN);  // (128, 16)
    gemm_wmma<half_t><<<ggrid, 256, 0, stream>>>(H16, Wqh, bq, Qh);
    gemm_wmma<half_t><<<ggrid, 256, 0, stream>>>(H16, Wkh, bk, Kh);
    gemm_wmma<half_t><<<ggrid, 256, 0, stream>>>(H16, Wvh, bv, Vh);

    // 4. Attention (one wave per (b,h); 8 waves per block)
    attn_kernel<<<(B_DIM * NH) / 8, 256, 0, stream>>>(Qh, Kh, Vh, quality, Ctxh);

    // 5. Output projection GEMM (f32 output, bias bo fused)
    gemm_wmma<float><<<ggrid, 256, 0, stream>>>(Ctxh, Woh, bo, CtxP);

    // 6. Fuse means + residual + LayerNorm
    finalize_kernel<<<B_DIM, 256, 0, stream>>>(CtxP, e0, e1, e2, e3, gamma, beta, out);

    (void)in_sizes; (void)n_in; (void)out_size; (void)ws_size;
}
